// GViTEncoder_21208548508277
// MI455X (gfx1250) — matrix operand order repaired and checked by reading, not compiled, not run on hardware
//
#include <hip/hip_runtime.h>
#include <hip/hip_bf16.h>

// ---------------------------------------------------------------------------
// GViT encoder block, fully fused for MI455X (gfx1250, wave32, WMMA + TDM).
// B=8, N=1024, F_IN=384, F_HID=96, HEADS=8, F_OUT=384.
// Dominant compute: A@V = 64 x [1024x1024]@[1024x96] = 12.9 GFLOP -> bf16 WMMA.
// A (softmax over heads, per (i,j)) is never materialized in HBM.
// Packed-V B-operands are streamed into LDS once per workgroup by the
// Tensor Data Mover (double-buffered, TENSORcnt), not 8x per-wave from global.
// ---------------------------------------------------------------------------

typedef __bf16 bf16_t;
typedef __attribute__((ext_vector_type(16))) __bf16 v16bf;
typedef __attribute__((ext_vector_type(8)))  __bf16 v8bf;
typedef __attribute__((ext_vector_type(8)))  float  v8f;
typedef unsigned int u32;
typedef __attribute__((ext_vector_type(4))) u32 v4u;
typedef __attribute__((ext_vector_type(8))) int v8i;
typedef __attribute__((ext_vector_type(4))) int v4i;

#define BATCH  8
#define SEQ    1024
#define FIN    384
#define FHID   96
#define NHEADS 8
#define FOUT   384
#define HDIM   (NHEADS * FHID + FIN)   // 1152
#define ODIM   (FOUT + HDIM)           // 1536
#define ROWS   (BATCH * SEQ)           // 8192
#define NEG_SLOPE 0.01f
#define LN_EPS    1e-5f

#if defined(__HIP_DEVICE_COMPILE__) && __has_builtin(__builtin_amdgcn_tensor_load_to_lds)
#define USE_TDM 1
#else
#define USE_TDM 0
#endif

// --- A-fragment (16xK tile, bf16, row-major in LDS), contiguous k order:
// lane<16 : M=lane,    K = 0..15
// lane>=16: M=lane-16, K = 16..31.  Two ds_read_b128 per lane.
__device__ inline v16bf load_a_frag(const bf16_t* rowbase, int ldm, int lane) {
  const int m  = lane & 15;
  const int hi = lane >> 4;
  const bf16_t* p = rowbase + m * ldm + hi * 16;
  v8bf lo = *(const v8bf*)(p);
  v8bf hh = *(const v8bf*)(p + 8);
  v16bf a;
#pragma unroll
  for (int i = 0; i < 8; ++i) { a[i] = lo[i]; a[8 + i] = hh[i]; }
  return a;
}

// Packed B-operand tile: 512 bf16 per (kt,nt) tile, lane-contiguous (32B/lane).
__device__ inline v16bf load_b_frag(const bf16_t* packed_tile, int lane) {
  return ((const v16bf*)packed_tile)[lane];
}

__device__ inline v8f wmma_bf16(v16bf a, v16bf b, v8f c) {
  return __builtin_amdgcn_wmma_f32_16x16x32_bf16(false, a, false, b,
                                                 (short)0, c, false, false);
}

__device__ inline float lrelu(float x) { return x >= 0.f ? x : NEG_SLOPE * x; }

#if USE_TDM
// ---------------------------------------------------------------------------
// TDM: 2D tile load (tile_d0 units of 8B per row, tile_d1 rows, row stride
// d0_stride units) from global into contiguous LDS at lds_addr. D# per ISA §8.
// ---------------------------------------------------------------------------
__device__ inline void tdm_load_2d(u32 lds_addr, const void* gaddr,
                                   u32 tile_d0, u32 tile_d1,
                                   u32 tens_d0, u32 tens_d1, u32 d0_stride) {
  unsigned long long ga = (unsigned long long)(uintptr_t)gaddr;
  v4u g0;
  g0[0] = 1u;                                   // count=1, user descriptor
  g0[1] = lds_addr;                             // LDS byte address
  g0[2] = (u32)(ga & 0xffffffffu);              // global_addr[31:0]
  g0[3] = (u32)((ga >> 32) & 0x01ffffffu) | (2u << 30);  // addr[56:32] | type=2
  v8i g1;
  g1[0] = (int)(3u << 16);                                  // data_size=3 (8B)
  g1[1] = (int)((tens_d0 & 0xffffu) << 16);                 // tensor_dim0 lo
  g1[2] = (int)((tens_d0 >> 16) & 0xffffu) | (int)((tens_d1 & 0xffffu) << 16);
  g1[3] = (int)((tens_d1 >> 16) & 0xffffu) | (int)((tile_d0 & 0xffffu) << 16);
  g1[4] = (int)(tile_d1 & 0xffffu);                         // tile_dim1, dim2=0
  g1[5] = (int)d0_stride;                                   // dim0_stride[31:0]
  g1[6] = 0;
  g1[7] = 0;
  v4i z4 = {0, 0, 0, 0};
#if defined(__clang_major__) && __clang_major__ >= 23
  v8i z8 = {0, 0, 0, 0, 0, 0, 0, 0};
  __builtin_amdgcn_tensor_load_to_lds(g0, g1, z4, z4, z8, 0);
#else
  __builtin_amdgcn_tensor_load_to_lds(g0, g1, z4, z4, 0);
#endif
}
#endif  // USE_TDM

// ---------------------------------------------------------------------------
// Pack a row-major f32 [K x Ncols] weight into bf16 B-fragment tiles:
// layout [nt][kt][512]; lane l holds column n = l&15, k = (l>>4)*16 + i.
// ---------------------------------------------------------------------------
__global__ __launch_bounds__(256) void pack_b_kernel(const float* __restrict__ W,
                                                     bf16_t* __restrict__ out,
                                                     int K, int Ncols) {
  int idx = blockIdx.x * blockDim.x + threadIdx.x;
  int total = K * Ncols;
  if (idx >= total) return;
  int tile = idx >> 9;
  int r    = idx & 511;
  int lane = r >> 4;
  int i    = r & 15;
  int nkt  = K >> 5;
  int nt   = tile / nkt;
  int kt   = tile - nt * nkt;
  int n    = nt * 16 + (lane & 15);
  int k    = kt * 32 + (lane >> 4) * 16 + i;
  out[idx] = (bf16_t)W[k * Ncols + n];
}

// Pack V [ROWS x FHID] f32 into per-batch B-fragment tiles over j (=K dim):
// layout [b][nt(6)][kt(32)][512].
__global__ __launch_bounds__(256) void pack_v_kernel(const float* __restrict__ V,
                                                     bf16_t* __restrict__ out) {
  int idx = blockIdx.x * blockDim.x + threadIdx.x;
  if (idx >= ROWS * FHID) return;
  const int per_b = SEQ * FHID;          // 98304
  int b   = idx / per_b;
  int t   = idx - b * per_b;
  int tile = t >> 9;
  int r    = t & 511;
  int lane = r >> 4;
  int i    = r & 15;
  const int nkt = SEQ / 32;              // 32
  int nt = tile / nkt;
  int kt = tile - nt * nkt;
  int f  = nt * 16 + (lane & 15);
  int j  = kt * 32 + (lane >> 4) * 16 + i;
  out[idx] = (bf16_t)V[(size_t)(b * SEQ + j) * FHID + f];
}

// ---------------------------------------------------------------------------
// Kernel 1: per 16-row tile (one wave): LayerNorm(X) -> bf16 LDS,
// V = Hn@Wv + bv via WMMA, then Q/K projections; also stage X into H[:,768:].
// ---------------------------------------------------------------------------
__global__ __launch_bounds__(32) void ln_v_qk_kernel(
    const float* __restrict__ X, const bf16_t* __restrict__ Wvp,
    const float* __restrict__ bv, const float* __restrict__ Wq,
    const float* __restrict__ bq, const float* __restrict__ Wk,
    const float* __restrict__ bk, float* __restrict__ V,
    float* __restrict__ Q, float* __restrict__ Kd,
    float* __restrict__ H) {
  __shared__ __attribute__((aligned(16))) bf16_t hn[16][FIN + 8];  // bank spread
  __shared__ float vt[16][FHID + 4];
  const int lane = threadIdx.x;
  const int r0   = blockIdx.x * 16;

  // LayerNorm over 384, one row at a time (wave-cooperative reduction)
  for (int row = 0; row < 16; ++row) {
    const float* xr = X + (size_t)(r0 + row) * FIN;
    float s = 0.f, ss = 0.f;
    for (int c = lane; c < FIN; c += 32) { float x = xr[c]; s += x; ss += x * x; }
#pragma unroll
    for (int off = 16; off; off >>= 1) {
      s  += __shfl_xor(s, off, 32);
      ss += __shfl_xor(ss, off, 32);
    }
    float mu   = s * (1.f / FIN);
    float rstd = rsqrtf(ss * (1.f / FIN) - mu * mu + LN_EPS);
    for (int c = lane; c < FIN; c += 32) {
      float x = xr[c];
      hn[row][c] = (bf16_t)((x - mu) * rstd);
      H[(size_t)(r0 + row) * HDIM + NHEADS * FHID + c] = x;  // residual concat
    }
  }
  __syncthreads();

  // V = Hn @ Wv : K=384 (12 k-tiles), N=96 (6 n-tiles)
  v8f acc[6];
#pragma unroll
  for (int t = 0; t < 6; ++t) acc[t] = (v8f){0.f,0.f,0.f,0.f,0.f,0.f,0.f,0.f};
  for (int kt = 0; kt < 12; ++kt) {
    v16bf a = load_a_frag(&hn[0][0] + kt * 32, FIN + 8, lane);
#pragma unroll
    for (int nt = 0; nt < 6; ++nt) {
      v16bf b = load_b_frag(Wvp + (size_t)(nt * 12 + kt) * 512, lane);
      acc[nt] = wmma_bf16(a, b, acc[nt]);
    }
  }
  const int nl = lane & 15;
  const int rh = (lane >> 4) * 8;
#pragma unroll
  for (int nt = 0; nt < 6; ++nt) {
    int n = nt * 16 + nl;
    float bias = bv[n];
#pragma unroll
    for (int v = 0; v < 8; ++v) {
      float val = acc[nt][v] + bias;
      vt[rh + v][n] = val;
      V[(size_t)(r0 + rh + v) * FHID + n] = val;
    }
  }
  __syncthreads();

  // Q/K: 16 rows x 8 heads, 4 (row,head) dots of length 96 per lane
  for (int t = lane; t < 128; t += 32) {
    int row = t >> 3, h = t & 7;
    float q = bq[h], k = bk[h];
    for (int f = 0; f < FHID; ++f) {
      float v = vt[row][f];
      q += v * Wq[f * NHEADS + h];
      k += v * Wk[f * NHEADS + h];
    }
    Q [(size_t)(r0 + row) * NHEADS + h] = q;
    Kd[(size_t)(r0 + row) * NHEADS + h] = k;
  }
}

// ---------------------------------------------------------------------------
// Kernel 2 (dominant): attention. Block = 8 waves; wave w owns head w.
// Per 32-j chunk: 256 threads cooperatively build A[i,j,h] =
// softmax_h(lrelu(Q_ih*K_jh)) in LDS (bf16), then each wave runs 6 WMMAs
// (A 16x32 @ Vpacked 32x16) accumulating Hh[i, h, 0:96] in f32.
// The 6 KB of packed-V B-tiles per chunk are DMA'd into LDS ONCE per block
// by the Tensor Data Mover, double buffered, overlapped with the VALU phase.
// ---------------------------------------------------------------------------
__global__ __launch_bounds__(256) void attn_kernel(
    const float* __restrict__ Q, const float* __restrict__ Kmat,
    const bf16_t* __restrict__ Vp, float* __restrict__ H) {
  const int b    = blockIdx.x >> 6;
  const int i0   = (blockIdx.x & 63) * 16;
  const int tid  = threadIdx.x;
  const int wave = tid >> 5;
  const int lane = tid & 31;

  __shared__ float  qs[16][NHEADS];
  __shared__ float  ks[32][NHEADS];
  __shared__ __attribute__((aligned(16))) bf16_t aT[NHEADS][16][40];  // padded
#if USE_TDM
  __shared__ __attribute__((aligned(16))) bf16_t vtile[2][6][512];    // 12 KB
#endif

  if (tid < 128)
    qs[tid >> 3][tid & 7] = Q[(size_t)(b * SEQ + i0 + (tid >> 3)) * NHEADS + (tid & 7)];

  v8f acc[6];
#pragma unroll
  for (int t = 0; t < 6; ++t) acc[t] = (v8f){0.f,0.f,0.f,0.f,0.f,0.f,0.f,0.f};

  const bf16_t* vbase = Vp + (size_t)b * (SEQ * FHID);

#if USE_TDM
  const u32 lds0 = (u32)(uintptr_t)&vtile[0][0][0];
  // prime the pipe: chunk 0 -> buffer 0
  if (wave == 0)
    tdm_load_2d(lds0, vbase, /*tile*/128, 6, /*tensor*/4096, 16, /*stride*/4096);
#endif
  __syncthreads();

  for (int jc = 0; jc < 32; ++jc) {
    // stage K chunk (32 rows x 8 heads), one element per thread
    ks[tid >> 3][tid & 7] =
        Kmat[(size_t)(b * SEQ + jc * 32 + (tid >> 3)) * NHEADS + (tid & 7)];
#if USE_TDM
    // async prefetch of next chunk's 6 B-tiles (wraps at the end so the
    // tensorcnt wait below can use a constant threshold)
    if (wave == 0) {
      int jn = (jc + 1) & 31;
      tdm_load_2d(lds0 + (u32)(((jc + 1) & 1) * sizeof(vtile[0])),
                  vbase + (size_t)jn * 512, 128, 6, 4096, 16, 4096);
    }
#else
    if (jc + 1 < 32)
      __builtin_prefetch(vbase + (size_t)(jc + 1) * 512, 0, 1);
#endif
    __syncthreads();

    // scores + softmax over heads: thread t -> (i = t/16, j = t%16 and +16)
    const int i  = tid >> 4;
    const int jl = tid & 15;
#pragma unroll
    for (int jj = 0; jj < 2; ++jj) {
      const int j = jl + jj * 16;
      float e[NHEADS], denom = 0.f;
#pragma unroll
      for (int h = 0; h < NHEADS; ++h) {
        float s = lrelu(qs[i][h] * ks[j][h]);
        e[h] = __expf(s);
        denom += e[h];
      }
      float inv = 1.f / denom;
#pragma unroll
      for (int h = 0; h < NHEADS; ++h) aT[h][i][j] = (bf16_t)(e[h] * inv);
    }
#if USE_TDM
    if (wave == 0) __builtin_amdgcn_s_wait_tensorcnt(1);  // chunk jc landed
#endif
    __syncthreads();

    // wave = head: A (16x32) @ packed V (32x16) for 6 n-tiles
    v16bf a = load_a_frag(&aT[wave][0][0], 40, lane);
#pragma unroll
    for (int nt = 0; nt < 6; ++nt) {
#if USE_TDM
      v16bf bfr = load_b_frag(&vtile[jc & 1][nt][0], lane);
#else
      v16bf bfr = load_b_frag(vbase + (size_t)(nt * 32 + jc) * 512, lane);
#endif
      acc[nt] = wmma_bf16(a, bfr, acc[nt]);
    }
    __syncthreads();  // protect ks/aT/vtile before next chunk
  }

  // write Hh into H[:, wave*96 + ...]
  const int nl = lane & 15;
  const int rh = (lane >> 4) * 8;
#pragma unroll
  for (int nt = 0; nt < 6; ++nt)
#pragma unroll
    for (int v = 0; v < 8; ++v)
      H[(size_t)(b * SEQ + i0 + rh + v) * HDIM + wave * FHID + nt * 16 + nl] =
          acc[nt][v];
}

// ---------------------------------------------------------------------------
// Kernel 3: O = lrelu(LN(H) @ Wm + bm); out = [O | H].
// Block = 4 waves over a 16-row tile; wave w owns n-tiles w*6..w*6+5 (N=384).
// K = 1152 -> 36 k-tiles; Hn staged in LDS bf16 (36 KB, padded rows).
// ---------------------------------------------------------------------------
__global__ __launch_bounds__(128) void out_kernel(
    const float* __restrict__ H, const bf16_t* __restrict__ Wmp,
    const float* __restrict__ bm, float* __restrict__ out) {
  __shared__ __attribute__((aligned(16))) bf16_t hn[16][HDIM + 8];
  const int r0   = blockIdx.x * 16;
  const int tid  = threadIdx.x;
  const int wave = tid >> 5;
  const int lane = tid & 31;

  // LayerNorm over 1152: wave w handles rows w, w+4, w+8, w+12
  for (int row = wave; row < 16; row += 4) {
    const float* hr = H + (size_t)(r0 + row) * HDIM;
    float s = 0.f, ss = 0.f;
    for (int c = lane; c < HDIM; c += 32) { float x = hr[c]; s += x; ss += x * x; }
#pragma unroll
    for (int off = 16; off; off >>= 1) {
      s  += __shfl_xor(s, off, 32);
      ss += __shfl_xor(ss, off, 32);
    }
    float mu   = s * (1.f / HDIM);
    float rstd = rsqrtf(ss * (1.f / HDIM) - mu * mu + LN_EPS);
    for (int c = lane; c < HDIM; c += 32)
      hn[row][c] = (bf16_t)((hr[c] - mu) * rstd);
  }
  __syncthreads();

  v8f acc[6];
#pragma unroll
  for (int t = 0; t < 6; ++t) acc[t] = (v8f){0.f,0.f,0.f,0.f,0.f,0.f,0.f,0.f};
  for (int kt = 0; kt < 36; ++kt) {
    v16bf a = load_a_frag(&hn[0][0] + kt * 32, HDIM + 8, lane);
#pragma unroll
    for (int t = 0; t < 6; ++t) {
      int nt = wave * 6 + t;
      v16bf b = load_b_frag(Wmp + (size_t)(nt * 36 + kt) * 512, lane);
      acc[t] = wmma_bf16(a, b, acc[t]);
    }
  }
  const int nl = lane & 15;
  const int rh = (lane >> 4) * 8;
#pragma unroll
  for (int t = 0; t < 6; ++t) {
    int n = (wave * 6 + t) * 16 + nl;
    float bias = bm[n];
#pragma unroll
    for (int v = 0; v < 8; ++v)
      out[(size_t)(r0 + rh + v) * ODIM + n] = lrelu(acc[t][v] + bias);
  }

  // out[:, 384:1536] = H
  for (int idx = tid; idx < 16 * HDIM; idx += 128) {
    int row = idx / HDIM, c = idx - row * HDIM;
    out[(size_t)(r0 + row) * ODIM + FOUT + c] = H[(size_t)(r0 + row) * HDIM + c];
  }
}

// ---------------------------------------------------------------------------
extern "C" void kernel_launch(void* const* d_in, const int* in_sizes, int n_in,
                              void* d_out, int out_size, void* d_ws, size_t ws_size,
                              hipStream_t stream) {
  const float* X  = (const float*)d_in[0];
  const float* Wv = (const float*)d_in[1];
  const float* bv = (const float*)d_in[2];
  const float* Wq = (const float*)d_in[3];
  const float* bq = (const float*)d_in[4];
  const float* Wk = (const float*)d_in[5];
  const float* bk = (const float*)d_in[6];
  const float* Wm = (const float*)d_in[7];
  const float* bm = (const float*)d_in[8];
  float* out = (float*)d_out;

  char* ws = (char*)d_ws;
  size_t off = 0;
  auto alloc = [&](size_t bytes) -> void* {
    void* p = ws + off;
    off = (off + bytes + 255) & ~(size_t)255;
    return p;
  };
  bf16_t* Wvp = (bf16_t*)alloc((size_t)FIN * FHID * 2);
  bf16_t* Wmp = (bf16_t*)alloc((size_t)HDIM * FOUT * 2);
  float*  V   = (float*) alloc((size_t)ROWS * FHID * 4);
  bf16_t* Vp  = (bf16_t*)alloc((size_t)ROWS * FHID * 2);
  float*  Q   = (float*) alloc((size_t)ROWS * NHEADS * 4);
  float*  Kd  = (float*) alloc((size_t)ROWS * NHEADS * 4);
  float*  H   = (float*) alloc((size_t)ROWS * HDIM * 4);
  (void)ws_size; (void)in_sizes; (void)n_in; (void)out_size;

  // 1) pack weights to bf16 B-fragment layout
  pack_b_kernel<<<(FIN * FHID + 255) / 256, 256, 0, stream>>>(Wv, Wvp, FIN, FHID);
  pack_b_kernel<<<(HDIM * FOUT + 255) / 256, 256, 0, stream>>>(Wm, Wmp, HDIM, FOUT);

  // 2) LN + V/Q/K projection (+ stage X into H residual columns)
  ln_v_qk_kernel<<<ROWS / 16, 32, 0, stream>>>(X, Wvp, bv, Wq, bq, Wk, bk,
                                               V, Q, Kd, H);

  // 3) pack V into per-batch B-fragment layout for A@V
  pack_v_kernel<<<(ROWS * FHID + 255) / 256, 256, 0, stream>>>(V, Vp);

  // 4) attention (softmax-over-heads fused into WMMA A@V, TDM-fed B tiles)
  attn_kernel<<<BATCH * (SEQ / 16), 256, 0, stream>>>(Q, Kd, Vp, H);

  // 5) output projection + concat
  out_kernel<<<ROWS / 16, 128, 0, stream>>>(H, Wmp, bm, out);
}
